// MultiLatentAttention_31001073943347
// MI455X (gfx1250) — compile-verified, matrix operand order checked by reading, not run on hardware
//
#include <hip/hip_runtime.h>
#include <hip/hip_bf16.h>

// ---------------------------------------------------------------------------
// MultiLatentAttention for MI455X (gfx1250).
// bf16 WMMA GEMMs with async-to-LDS double-buffered staging + fused aux ops.
// Key insight: causal mask (t <= l, l < 64) means only t<64 keys/values matter.
// ---------------------------------------------------------------------------

#define BB   4
#define TT   4096
#define CC   1024
#define HH   16
#define LL   64
#define DD   64
#define LEN_SCALE (1.0f/64.0f)       // 1/sqrt(T)
#define ATT_SCALE 0.125f             // 1/sqrt(D)
#define LN_THETA  9.210340371976184f // ln(10000)

#if defined(__HIP_DEVICE_COMPILE__) && \
    __has_builtin(__builtin_amdgcn_global_load_async_to_lds_b128) && \
    __has_builtin(__builtin_amdgcn_s_wait_asynccnt)
#define HAVE_ASYNC 1
#else
#define HAVE_ASYNC 0
#endif

typedef __attribute__((ext_vector_type(16))) __bf16 bf16x16;
typedef __attribute__((ext_vector_type(8)))  __bf16 bf16x8;
typedef __attribute__((ext_vector_type(8)))  float  f32x8;
typedef __attribute__((ext_vector_type(4)))  int    i32x4;

#define GPTR(p) ((__attribute__((address_space(1))) i32x4*)(p))
#define LPTR(p) ((__attribute__((address_space(3))) i32x4*)(p))

__device__ __forceinline__ unsigned short f2bf(float f) {
    unsigned u = __float_as_uint(f);
    u += 0x7FFFu + ((u >> 16) & 1u);        // round-to-nearest-even
    return (unsigned short)(u >> 16);
}

__device__ __forceinline__ void store4bf(unsigned short* p, float4 f) {
    unsigned lo = (unsigned)f2bf(f.x) | ((unsigned)f2bf(f.y) << 16);
    unsigned hi = (unsigned)f2bf(f.z) | ((unsigned)f2bf(f.w) << 16);
    *reinterpret_cast<uint2*>(p) = make_uint2(lo, hi);
}

// Fragment load from LDS: per-lane K pattern {kb..kb+7, kb+16..kb+23},
// kb = (lane<16 ? 0 : 8), matching the CDNA5 16-bit 16x32 A/B layout.
__device__ __forceinline__ bf16x16 load_frag(const unsigned short* p) {
    bf16x8 lo = *reinterpret_cast<const bf16x8*>(p);
    bf16x8 hi = *reinterpret_cast<const bf16x8*>(p + 16);
    union { bf16x16 v; bf16x8 h[2]; } u;
    u.h[0] = lo; u.h[1] = hi;
    return u.v;
}

// ---------------------------------------------------------------------------
// fp32 -> bf16 bulk convert (n multiple of 8).
// ---------------------------------------------------------------------------
__global__ __launch_bounds__(256)
void cvt_bf16(const float* __restrict__ s, unsigned short* __restrict__ d, int n)
{
    int i = (blockIdx.x * 256 + threadIdx.x) * 8;
    if (i >= n) return;
    float4 a = *reinterpret_cast<const float4*>(s + i);
    float4 b = *reinterpret_cast<const float4*>(s + i + 4);
    store4bf(d + i, a);
    store4bf(d + i + 4, b);
}

// ---------------------------------------------------------------------------
// Generic C = A @ W^T GEMM: bf16 operands in global memory, f32 accumulate,
// output f32 or bf16. Tiles staged to LDS with GLOBAL_LOAD_ASYNC_TO_LDS_B128
// (double-buffered, ASYNCcnt-pipelined) when available.
//   A: [M,K] bf16 rows @ lda (optional gather64: row r -> first-64 rows/batch)
//   W: [N,K] bf16 rows @ ldw.  grid.z slices: A,C via (z>>4,z&15) strides,
//   W via z*sWz.  Block tile (WM*32) x (WN*64), 8 waves, wave tile 32x64, K step 32.
// ---------------------------------------------------------------------------
template<int WM, int WN>
__global__ __launch_bounds__(256)
void gemm_bf16(const unsigned short* __restrict__ A, int lda, int gather64,
               long long sAb, long long sAh,
               const unsigned short* __restrict__ W, int ldw, long long sWz,
               void* __restrict__ Cv, int ldc, long long sCb, long long sCh,
               int K, int out_bf16)
{
    constexpr int MT = WM * 32;
    constexpr int NT = WN * 64;
    constexpr int SK = 40;                   // padded LDS row stride (bf16 units)
    constexpr int AOPS = (MT * 4) / 256;     // per-thread 16B chunks per A tile
    constexpr int BOPS = (NT * 4) / 256;
    __shared__ unsigned short sA[2][MT * SK];
    __shared__ unsigned short sB[2][NT * SK];

    const int tid  = threadIdx.x;
    const int lane = tid & 31;
    const int w    = tid >> 5;
    const int wm   = w / WN;
    const int wn   = w % WN;

    const int z = blockIdx.z;
    const long long zb = z >> 4, zh = z & 15;
    const unsigned short* Ab = A + zb * sAb + zh * sAh;
    const unsigned short* Wb = W + (long long)z * sWz;

    const int m0 = blockIdx.y * MT;
    const int n0 = blockIdx.x * NT;

    // Stage one 32-wide K tile (A and W) into LDS buffer `buf`.
    auto stage = [&](int kt, int buf) {
        const int k0 = kt * 32;
#pragma unroll
        for (int it = 0; it < AOPS; ++it) {
            int idx = tid + it * 256;
            int row = idx >> 2, ch = idx & 3;             // 4 x 8-bf16 chunks/row
            int gr  = m0 + row;
            long long srow = gather64 ? ((long long)(gr >> 6) * TT + (gr & 63))
                                      : (long long)gr;
            const unsigned short* gp = Ab + srow * lda + k0 + ch * 8;
            unsigned short*       lp = &sA[buf][row * SK + ch * 8];
#if HAVE_ASYNC
            __builtin_amdgcn_global_load_async_to_lds_b128(GPTR(gp), LPTR(lp), 0, 0);
#else
            *reinterpret_cast<uint4*>(lp) = *reinterpret_cast<const uint4*>(gp);
#endif
        }
#pragma unroll
        for (int it = 0; it < BOPS; ++it) {
            int idx = tid + it * 256;
            int row = idx >> 2, ch = idx & 3;
            const unsigned short* gp = Wb + (long long)(n0 + row) * ldw + k0 + ch * 8;
            unsigned short*       lp = &sB[buf][row * SK + ch * 8];
#if HAVE_ASYNC
            __builtin_amdgcn_global_load_async_to_lds_b128(GPTR(gp), LPTR(lp), 0, 0);
#else
            *reinterpret_cast<uint4*>(lp) = *reinterpret_cast<const uint4*>(gp);
#endif
        }
    };

    f32x8 zerov = {};
    f32x8 acc[2][4];
#pragma unroll
    for (int i = 0; i < 2; ++i)
#pragma unroll
        for (int j = 0; j < 4; ++j) acc[i][j] = zerov;

    const int nk = K >> 5;
    stage(0, 0);
    for (int kt = 0; kt < nk; ++kt) {
        const int buf = kt & 1;
        if (kt + 1 < nk) {
            stage(kt + 1, buf ^ 1);
#if HAVE_ASYNC
            __builtin_amdgcn_s_wait_asynccnt(AOPS + BOPS);  // tile kt retired
#endif
        } else {
#if HAVE_ASYNC
            __builtin_amdgcn_s_wait_asynccnt(0);
#endif
        }
        __syncthreads();                                    // tile kt visible

        const int kb = (lane >> 4) << 3;
        bf16x16 afr[2], bfr[4];
#pragma unroll
        for (int mi = 0; mi < 2; ++mi)
            afr[mi] = load_frag(&sA[buf][(wm * 32 + mi * 16 + (lane & 15)) * SK + kb]);
#pragma unroll
        for (int ni = 0; ni < 4; ++ni)
            bfr[ni] = load_frag(&sB[buf][(wn * 64 + ni * 16 + (lane & 15)) * SK + kb]);

#pragma unroll
        for (int mi = 0; mi < 2; ++mi)
#pragma unroll
            for (int ni = 0; ni < 4; ++ni)
                acc[mi][ni] = __builtin_amdgcn_wmma_f32_16x16x32_bf16(
                    false, afr[mi], false, bfr[ni], (short)0, acc[mi][ni],
                    false, false);
        __syncthreads();                                    // free buf for kt+2
    }

    // Epilogue. C 16x16 f32 layout: VGPR r, lanes 0-15 -> M=r, lanes 16-31 -> M=r+8.
    const int rbase = (lane >> 4) << 3;
    if (out_bf16) {
        unsigned short* Cb = (unsigned short*)Cv + zb * sCb + zh * sCh;
#pragma unroll
        for (int mi = 0; mi < 2; ++mi)
#pragma unroll
            for (int ni = 0; ni < 4; ++ni)
#pragma unroll
                for (int r = 0; r < 8; ++r) {
                    int mr = m0 + wm * 32 + mi * 16 + rbase + r;
                    int nc = n0 + wn * 64 + ni * 16 + (lane & 15);
                    Cb[(long long)mr * ldc + nc] = f2bf(acc[mi][ni][r]);
                }
    } else {
        float* Cb = (float*)Cv + zb * sCb + zh * sCh;
#pragma unroll
        for (int mi = 0; mi < 2; ++mi)
#pragma unroll
            for (int ni = 0; ni < 4; ++ni)
#pragma unroll
                for (int r = 0; r < 8; ++r) {
                    int mr = m0 + wm * 32 + mi * 16 + rbase + r;
                    int nc = n0 + wn * 64 + ni * 16 + (lane & 15);
                    Cb[(long long)mr * ldc + nc] = acc[mi][ni][r];
                }
    }
}

// ---------------------------------------------------------------------------
// K/V postprocess: per-(b,t<64,h) row of 64: layernorm * len_scale; RoPE on K.
// One wave per row; lane holds d=lane and d=lane+32.
// ---------------------------------------------------------------------------
__global__ __launch_bounds__(256)
void kv_post(float* __restrict__ Kb, float* __restrict__ Vb,
             const float* __restrict__ knw, const float* __restrict__ knb,
             const float* __restrict__ vnw, const float* __restrict__ vnb)
{
    int row  = blockIdx.x * 8 + (threadIdx.x >> 5);   // 0..4095  (b,t,h)
    int lane = threadIdx.x & 31;
    int h = row & 15, t = (row >> 4) & 63, b = row >> 10;
    size_t off = ((size_t)(b * 64 + t)) * CC + h * DD;

    float k0 = Kb[off + lane], k1 = Kb[off + lane + 32];
    float v0 = Vb[off + lane], v1 = Vb[off + lane + 32];

    auto lnorm = [&](float& a, float& c, const float* wt, const float* bi) {
        float s = a + c;
        for (int m = 16; m > 0; m >>= 1) s += __shfl_xor(s, m, 32);
        float mean = s * (1.0f / 64.0f);
        float d0 = a - mean, d1 = c - mean;
        float vv = d0 * d0 + d1 * d1;
        for (int m = 16; m > 0; m >>= 1) vv += __shfl_xor(vv, m, 32);
        float rstd = rsqrtf(vv * (1.0f / 64.0f) + 1e-5f);
        a = (d0 * rstd * wt[lane]      + bi[lane])      * LEN_SCALE;
        c = (d1 * rstd * wt[lane + 32] + bi[lane + 32]) * LEN_SCALE;
    };
    lnorm(k0, k1, knw, knb);
    lnorm(v0, v1, vnw, vnb);

    // RoPE on K: pair (2i, 2i+1); partner via lane-xor-1 within each half.
    float p0 = __shfl_xor(k0, 1, 32);
    float p1 = __shfl_xor(k1, 1, 32);
    int   i0 = lane >> 1;
    int   i1 = 16 + (lane >> 1);
    float f0 = __expf(-((float)(2 * i0) / 64.0f) * LN_THETA);
    float f1 = __expf(-((float)(2 * i1) / 64.0f) * LN_THETA);
    float c0 = __cosf(t * f0), s0 = __sinf(t * f0);
    float c1 = __cosf(t * f1), s1 = __sinf(t * f1);
    float r0 = ((lane & 1) == 0) ? (k0 * c0 - p0 * s0) : (p0 * s0 + k0 * c0);
    float r1 = ((lane & 1) == 0) ? (k1 * c1 - p1 * s1) : (p1 * s1 + k1 * c1);

    Kb[off + lane] = r0;  Kb[off + lane + 32] = r1;
    Vb[off + lane] = v0;  Vb[off + lane + 32] = v1;
}

// ---------------------------------------------------------------------------
// Gate softmax over L=64; one wave per (b,t,h) row; f32 in, bf16 out.
// ---------------------------------------------------------------------------
__global__ __launch_bounds__(256)
void gates_softmax(const float* __restrict__ G, unsigned short* __restrict__ Gbf)
{
    int row  = blockIdx.x * 8 + (threadIdx.x >> 5);
    int lane = threadIdx.x & 31;
    const float* g = G + (size_t)row * LL;
    unsigned short* o = Gbf + (size_t)row * LL;
    float a0 = g[lane] * ATT_SCALE, a1 = g[lane + 32] * ATT_SCALE;
    float mx = fmaxf(a0, a1);
    for (int m = 16; m > 0; m >>= 1) mx = fmaxf(mx, __shfl_xor(mx, m, 32));
    float e0 = __expf(a0 - mx), e1 = __expf(a1 - mx);
    float s = e0 + e1;
    for (int m = 16; m > 0; m >>= 1) s += __shfl_xor(s, m, 32);
    float inv = 1.0f / s;
    o[lane] = f2bf(e0 * inv);  o[lane + 32] = f2bf(e1 * inv);
}

// ---------------------------------------------------------------------------
// Latent attention per (b,h): q=LN(latents)*ls; scores=q@k^T*scale, causal
// (t<=l), softmax, out = attn @ v.  Writes attn_out transposed [b,h,d,l] bf16.
// ---------------------------------------------------------------------------
__global__ __launch_bounds__(256)
void mla_attn(const float* __restrict__ lq,
              const float* __restrict__ Kb, const float* __restrict__ Vb,
              const float* __restrict__ qnw, const float* __restrict__ qnb,
              unsigned short* __restrict__ AOT)
{
    __shared__ float sq[LL * DD];
    __shared__ float sk[LL * DD];
    __shared__ float sv[LL * DD];
    __shared__ float sa[8 * LL];

    int b = blockIdx.x >> 4, h = blockIdx.x & 15;
    int tid = threadIdx.x, lane = tid & 31, w = tid >> 5;

    for (int i = tid; i < LL * DD; i += 256) {
        int t = i >> 6, d = i & 63;
        size_t off = ((size_t)(b * 64 + t)) * CC + h * DD + d;
        sk[i] = Kb[off];
        sv[i] = Vb[off];
    }
    for (int l = w; l < LL; l += 8) {
        float q0 = lq[(l * HH + h) * DD + lane];
        float q1 = lq[(l * HH + h) * DD + lane + 32];
        float s = q0 + q1;
        for (int m = 16; m > 0; m >>= 1) s += __shfl_xor(s, m, 32);
        float mean = s * (1.0f / 64.0f);
        float d0 = q0 - mean, d1 = q1 - mean;
        float vv = d0 * d0 + d1 * d1;
        for (int m = 16; m > 0; m >>= 1) vv += __shfl_xor(vv, m, 32);
        float rstd = rsqrtf(vv * (1.0f / 64.0f) + 1e-5f);
        sq[l * DD + lane]      = (d0 * rstd * qnw[lane]      + qnb[lane])      * LEN_SCALE;
        sq[l * DD + lane + 32] = (d1 * rstd * qnw[lane + 32] + qnb[lane + 32]) * LEN_SCALE;
    }
    __syncthreads();

    for (int l = w; l < LL; l += 8) {
        float s0 = 0.0f, s1 = 0.0f;      // scores for t = lane, lane+32
        for (int d = 0; d < DD; ++d) {
            float q = sq[l * DD + d];
            s0 += q * sk[lane * DD + d];
            s1 += q * sk[(lane + 32) * DD + d];
        }
        s0 *= ATT_SCALE;  s1 *= ATT_SCALE;
        if (lane > l)      s0 = -1e30f;
        if (lane + 32 > l) s1 = -1e30f;
        float mx = fmaxf(s0, s1);
        for (int m = 16; m > 0; m >>= 1) mx = fmaxf(mx, __shfl_xor(mx, m, 32));
        float e0 = __expf(s0 - mx), e1 = __expf(s1 - mx);
        float se = e0 + e1;
        for (int m = 16; m > 0; m >>= 1) se += __shfl_xor(se, m, 32);
        float inv = 1.0f / se;
        sa[w * LL + lane]      = e0 * inv;
        sa[w * LL + lane + 32] = e1 * inv;      // same-wave LDS RAW: in-order
        float o0 = 0.0f, o1 = 0.0f;
        for (int t = 0; t <= l; ++t) {
            float a = sa[w * LL + t];
            o0 += a * sv[t * DD + lane];
            o1 += a * sv[t * DD + lane + 32];
        }
        size_t base = ((size_t)(b * HH + h)) * (DD * LL);
        AOT[base + lane * LL + l]        = f2bf(o0);   // transposed: [d][l]
        AOT[base + (lane + 32) * LL + l] = f2bf(o1);
    }
}

// ---------------------------------------------------------------------------
extern "C" void kernel_launch(void* const* d_in, const int* in_sizes, int n_in,
                              void* d_out, int out_size, void* d_ws, size_t ws_size,
                              hipStream_t stream)
{
    const float* x   = (const float*)d_in[0];
    const float* lq  = (const float*)d_in[1];
    const float* wk  = (const float*)d_in[2];
    const float* wv  = (const float*)d_in[3];
    const float* wg  = (const float*)d_in[4];
    const float* wp  = (const float*)d_in[5];
    const float* qnw = (const float*)d_in[6];
    const float* qnb = (const float*)d_in[7];
    const float* knw = (const float*)d_in[8];
    const float* knb = (const float*)d_in[9];
    const float* vnw = (const float*)d_in[10];
    const float* vnb = (const float*)d_in[11];
    float* out = (float*)d_out;

    const size_t NX = (size_t)BB * TT * CC;        // 16,777,216
    const size_t NW = (size_t)CC * CC;             //  1,048,576
    const size_t NA = (size_t)BB * HH * DD * LL;   //    262,144
    const size_t NKV = (size_t)BB * 64 * CC;       //    262,144

    // f32 regions first, then bf16 regions (all 16B aligned).
    float* ws  = (float*)d_ws;
    float* Kb  = ws;                 // [B*64, C]
    float* Vb  = Kb + NKV;
    float* Gb  = Vb + NKV;           // [B*T, C] pre-softmax gates (f32)
    unsigned short* xb   = (unsigned short*)(Gb + NX);
    unsigned short* wkb  = xb   + NX;
    unsigned short* wvb  = wkb  + NW;
    unsigned short* wgb  = wvb  + NW;
    unsigned short* wpb  = wgb  + NW;
    unsigned short* Gbf  = wpb  + NW;   // softmaxed gates (bf16)
    unsigned short* AOTb = Gbf  + NX;   // attn_out^T [b,h,d,l] (bf16)
    unsigned short* Wtb  = AOTb + NA;   // weighted [b,t,h,d] (bf16)

    // 0) one-time bf16 conversions of GEMM operands.
    cvt_bf16<<<dim3(NX / 2048), 256, 0, stream>>>(x,  xb,  (int)NX);
    cvt_bf16<<<dim3(NW / 2048), 256, 0, stream>>>(wk, wkb, (int)NW);
    cvt_bf16<<<dim3(NW / 2048), 256, 0, stream>>>(wv, wvb, (int)NW);
    cvt_bf16<<<dim3(NW / 2048), 256, 0, stream>>>(wg, wgb, (int)NW);
    cvt_bf16<<<dim3(NW / 2048), 256, 0, stream>>>(wp, wpb, (int)NW);

    // 1-3) Projections: K (64 rows/batch, gathered), V, gates (full T).
    gemm_bf16<4, 2><<<dim3(8, 2, 1), 256, 0, stream>>>(
        xb, CC, 1, 0, 0, wkb, CC, 0, Kb, CC, 0, 0, CC, 0);
    gemm_bf16<4, 2><<<dim3(8, 2, 1), 256, 0, stream>>>(
        xb, CC, 1, 0, 0, wvb, CC, 0, Vb, CC, 0, 0, CC, 0);
    gemm_bf16<4, 2><<<dim3(8, 128, 1), 256, 0, stream>>>(
        xb, CC, 0, 0, 0, wgb, CC, 0, Gb, CC, 0, 0, CC, 0);

    // 4) layernorm + length scale + RoPE on K; layernorm + scale on V.
    kv_post<<<dim3(512), 256, 0, stream>>>(Kb, Vb, knw, knb, vnw, vnb);

    // 5) gate softmax over L -> bf16.
    gates_softmax<<<dim3((BB * TT * HH) / 8), 256, 0, stream>>>(Gb, Gbf);

    // 6) latent attention -> attn_out^T [b,h,d,l] bf16.
    mla_attn<<<dim3(BB * HH), 256, 0, stream>>>(lq, Kb, Vb, qnw, qnb, AOTb);

    // 7) weighted[b,t,h,d] = gates[b,t,h,:] @ attn_out[b,h,:,d]  (K=L=64),
    //    bf16 output for the final GEMM.
    gemm_bf16<8, 1><<<dim3(1, 16, BB * HH), 256, 0, stream>>>(
        Gbf, CC, 0, (long long)TT * HH * LL, (long long)LL,
        AOTb, LL, (long long)DD * LL,
        Wtb, CC, (long long)TT * HH * DD, (long long)DD,
        LL, 1);

    // 8) out = weighted @ wp^T (f32 out).
    gemm_bf16<4, 2><<<dim3(8, 128, 1), 256, 0, stream>>>(
        Wtb, CC, 0, 0, 0, wpb, CC, 0, out, CC, 0, 0, CC, 0);
}
